// AdditiveCoupling_40123584479654
// MI455X (gfx1250) — matrix operand order repaired and checked by reading, not compiled, not run on hardware
//
#include <hip/hip_runtime.h>
#include <hip/hip_bf16.h>

typedef __bf16 v16bf __attribute__((ext_vector_type(16)));
typedef float  v8f   __attribute__((ext_vector_type(8)));

#define B_ROWS   4096
#define D_DIM    2048
#define HALF_DIM 1024
#define MID_DIM  4096
#define NH       4

union Frag32B { v16bf v; uint4 q[2]; };

// ---- CDNA5 async global->LDS copy (16B per lane), tracked by ASYNCcnt ----
__device__ __forceinline__ void async_cp16(const void* lds_dst, const void* gsrc) {
  // Generic LDS pointer low 32 bits == wave-relative LDS byte offset.
  unsigned l = (unsigned)(uintptr_t)lds_dst;
  unsigned long long g = (unsigned long long)(uintptr_t)gsrc;
  asm volatile("global_load_async_to_lds_b128 %0, %1, off"
               :: "v"(l), "v"(g) : "memory");
}

template <int N>
__device__ __forceinline__ void wait_asynccnt() {
#if __has_builtin(__builtin_amdgcn_s_wait_asynccnt)
  __builtin_amdgcn_s_wait_asynccnt((unsigned short)N);
#else
  asm volatile("s_wait_asynccnt %0" :: "i"(N) : "memory");
#endif
}

// Deinterleave: x2 -> bf16 activations, odd output lanes pass through.
__global__ __launch_bounds__(256) void prep_kernel(const float* __restrict__ x,
                                                   __bf16* __restrict__ x2,
                                                   float* __restrict__ out) {
  int i = blockIdx.x * 256 + threadIdx.x;           // over B*HALF
  float2 p = reinterpret_cast<const float2*>(x)[i]; // (x1, x2) pair
  x2[i] = (__bf16)p.y;
  out[2 * (size_t)i + 1] = p.y;                     // y odd lanes = x2 (unchanged)
}

__global__ void ldj_kernel(const float* __restrict__ s, float* __restrict__ out) {
  if (threadIdx.x == 0) out[(size_t)B_ROWS * D_DIM] = s[0];
}

// W (K x N, f32) -> Wt (N x K, bf16), LDS-tiled transpose.
__global__ __launch_bounds__(256) void transpose_bf16_kernel(const float* __restrict__ W,
                                                             __bf16* __restrict__ Wt,
                                                             int K, int N) {
  __shared__ float tile[32][33];
  int n = blockIdx.x * 32 + threadIdx.x;
  int k = blockIdx.y * 32 + threadIdx.y;
  #pragma unroll
  for (int j = 0; j < 32; j += 8)
    tile[threadIdx.y + j][threadIdx.x] = W[(size_t)(k + j) * N + n];
  __syncthreads();
  int ko = blockIdx.y * 32 + threadIdx.x;
  int no = blockIdx.x * 32 + threadIdx.y;
  #pragma unroll
  for (int j = 0; j < 32; j += 8)
    Wt[(size_t)(no + j) * K + ko] = (__bf16)tile[threadIdx.x][threadIdx.y + j];
}

// C = act(A(MxK,bf16) * Bt(NxK,bf16)^T + bias)
// MODE 0: bf16 output with ReLU (hidden layers)
// MODE 1: f32 interleaved output: out[row*D + 2*col] = x[row*D + 2*col] + acc + bias
template <int MODE>
__global__ __launch_bounds__(256) void gemm_bf16_kernel(
    const __bf16* __restrict__ A, const __bf16* __restrict__ Bt,
    const float* __restrict__ bias, void* __restrict__ Cout,
    const float* __restrict__ xres, int M, int N, int K) {
  constexpr int BK  = 64;
  constexpr int LDP = BK + 8;     // 144B row stride: 16B aligned, conflict-free banks
  __shared__ __bf16 As[2][128][LDP];
  __shared__ __bf16 Bs[2][128][LDP];

  const int tid  = threadIdx.x;
  const int lane = tid & 31;
  const int wave = tid >> 5;
  const int lh   = lane & 15;
  const int hi   = lane >> 4;     // 0: lanes 0-15, 1: lanes 16-31

  const int m0  = blockIdx.y * 128;
  const int n0  = blockIdx.x * 128;
  const int wm0 = (wave & 1) * 64;    // 2x4 wave grid, 64x32 per wave
  const int wn0 = (wave >> 1) * 32;

  v8f acc[4][2];
  #pragma unroll
  for (int i = 0; i < 4; i++)
    #pragma unroll
    for (int j = 0; j < 2; j++) acc[i][j] = {};

  // Staging: 256 threads x 16B x 4 passes per matrix. row lr+32p, col lc.
  const int lr = tid >> 3;          // 0..31
  const int lc = (tid & 7) * 8;     // element offset 0..56
  const __bf16* Ag = A  + (size_t)(m0 + lr) * K + lc;
  const __bf16* Bg = Bt + (size_t)(n0 + lr) * K + lc;

  const int nk = K / BK;
  const int kA = hi ? 8 : 0;        // A frag: lanes16-31 hold K 8..15,24..31
  const int kB = hi ? 8 : 0;        // B frag: lanes16-31 hold K 8..15,24..31

  // Prologue: async-stage tile 0 into buffer 0.
  {
    #pragma unroll
    for (int p = 0; p < 4; ++p) {
      async_cp16(&As[0][lr + 32 * p][lc], Ag + (size_t)(32 * p) * K);
      async_cp16(&Bs[0][lr + 32 * p][lc], Bg + (size_t)(32 * p) * K);
    }
  }

  for (int kt = 0; kt < nk; ++kt) {
    const int cur = kt & 1;
    if (kt + 1 < nk) {              // async-stage next tile into the other buffer
      const __bf16* An = Ag + (size_t)(kt + 1) * BK;
      const __bf16* Bn = Bg + (size_t)(kt + 1) * BK;
      #pragma unroll
      for (int p = 0; p < 4; ++p) {
        async_cp16(&As[cur ^ 1][lr + 32 * p][lc], An + (size_t)(32 * p) * K);
        async_cp16(&Bs[cur ^ 1][lr + 32 * p][lc], Bn + (size_t)(32 * p) * K);
      }
      wait_asynccnt<8>();           // current tile's 8 copies are done
    } else {
      wait_asynccnt<0>();
    }
    __syncthreads();                // all waves' copies visible

    #pragma unroll
    for (int ks = 0; ks < 2; ++ks) {
      const int kA2 = ks * 32 + kA;
      const int kB2 = ks * 32 + kB;
      Frag32B a[4], b[2];
      #pragma unroll
      for (int i = 0; i < 4; i++) {
        const __bf16* p = &As[cur][wm0 + i * 16 + lh][0];
        a[i].q[0] = *reinterpret_cast<const uint4*>(p + kA2);      // K 0..7  / 8..15
        a[i].q[1] = *reinterpret_cast<const uint4*>(p + kA2 + 16); // K 16..23/24..31
      }
      #pragma unroll
      for (int j = 0; j < 2; j++) {
        const __bf16* p = &Bs[cur][wn0 + j * 16 + lh][0];
        b[j].q[0] = *reinterpret_cast<const uint4*>(p + kB2);      // K 0..7  / 8..15
        b[j].q[1] = *reinterpret_cast<const uint4*>(p + kB2 + 16); // K 16..23/24..31
      }
      #pragma unroll
      for (int i = 0; i < 4; i++)
        #pragma unroll
        for (int j = 0; j < 2; j++)
          acc[i][j] = __builtin_amdgcn_wmma_f32_16x16x32_bf16(
              false, a[i].v, false, b[j].v, (short)0, acc[i][j], false, false);
    }

    __syncthreads();                // everyone done reading buf[cur] before overwrite
  }

  // Epilogue. C/D layout: VGPR v -> row v (lanes 0-15) or row 8+v (lanes 16-31).
  #pragma unroll
  for (int j = 0; j < 2; j++) {
    const int col = n0 + wn0 + j * 16 + lh;
    const float bv = bias[col];
    #pragma unroll
    for (int i = 0; i < 4; i++) {
      #pragma unroll
      for (int v = 0; v < 8; v++) {
        const int row = m0 + wm0 + i * 16 + hi * 8 + v;
        float r = acc[i][j][v] + bv;
        if (MODE == 0) {
          r = r > 0.f ? r : 0.f;
          reinterpret_cast<__bf16*>(Cout)[(size_t)row * N + col] = (__bf16)r;
        } else {
          const size_t idx = (size_t)row * D_DIM + 2 * (size_t)col;
          reinterpret_cast<float*>(Cout)[idx] = xres[idx] + r;   // x1 + out
        }
      }
    }
  }
}

extern "C" void kernel_launch(void* const* d_in, const int* in_sizes, int n_in,
                              void* d_out, int out_size, void* d_ws, size_t ws_size,
                              hipStream_t stream) {
  const float* x     = (const float*)d_in[0];
  const float* ldj   = (const float*)d_in[1];
  const float* W_in  = (const float*)d_in[2];
  const float* b_in  = (const float*)d_in[3];
  const float* W_h   = (const float*)d_in[4];
  const float* b_h   = (const float*)d_in[5];
  const float* W_out = (const float*)d_in[6];
  const float* b_out = (const float*)d_in[7];
  float* out = (float*)d_out;

  char* ws = (char*)d_ws;                                 // ~216 MB used
  __bf16* x2    = (__bf16*)(ws);                          //   8 MB
  __bf16* hA    = (__bf16*)(ws + (8ull   << 20));         //  32 MB
  __bf16* hB    = (__bf16*)(ws + (40ull  << 20));         //  32 MB
  __bf16* WinT  = (__bf16*)(ws + (72ull  << 20));         //   8 MB
  __bf16* WhT   = (__bf16*)(ws + (80ull  << 20));         // 128 MB
  __bf16* WoutT = (__bf16*)(ws + (208ull << 20));         //   8 MB

  prep_kernel<<<(B_ROWS * HALF_DIM) / 256, 256, 0, stream>>>(x, x2, out);
  ldj_kernel<<<1, 32, 0, stream>>>(ldj, out);

  transpose_bf16_kernel<<<dim3(MID_DIM / 32, HALF_DIM / 32), dim3(32, 8), 0, stream>>>(
      W_in, WinT, HALF_DIM, MID_DIM);
  for (int i = 0; i < NH; i++)
    transpose_bf16_kernel<<<dim3(MID_DIM / 32, MID_DIM / 32), dim3(32, 8), 0, stream>>>(
        W_h + (size_t)i * MID_DIM * MID_DIM, WhT + (size_t)i * MID_DIM * MID_DIM,
        MID_DIM, MID_DIM);
  transpose_bf16_kernel<<<dim3(HALF_DIM / 32, MID_DIM / 32), dim3(32, 8), 0, stream>>>(
      W_out, WoutT, MID_DIM, HALF_DIM);

  // h = relu(x2 @ W_in + b_in)
  gemm_bf16_kernel<0><<<dim3(MID_DIM / 128, B_ROWS / 128), 256, 0, stream>>>(
      x2, WinT, b_in, hA, nullptr, B_ROWS, MID_DIM, HALF_DIM);

  __bf16* hin = hA; __bf16* hout = hB;
  for (int i = 0; i < NH; i++) {
    gemm_bf16_kernel<0><<<dim3(MID_DIM / 128, B_ROWS / 128), 256, 0, stream>>>(
        hin, WhT + (size_t)i * MID_DIM * MID_DIM, b_h + (size_t)i * MID_DIM,
        hout, nullptr, B_ROWS, MID_DIM, MID_DIM);
    __bf16* t = hin; hin = hout; hout = t;
  }

  // y_even = x1 + (h @ W_out + b_out), interleaved into d_out
  gemm_bf16_kernel<1><<<dim3(HALF_DIM / 128, B_ROWS / 128), 256, 0, stream>>>(
      hin, WoutT, b_out, out, x, B_ROWS, HALF_DIM, MID_DIM);

  (void)in_sizes; (void)n_in; (void)out_size; (void)ws_size;
}
